// NoSegAugmentedMemoryTransformer_12249246728831
// MI455X (gfx1250) — compile-verified, matrix operand order checked by reading, not run on hardware
//
#include <hip/hip_runtime.h>

// ---------------- problem constants ----------------
#define T_LEN   1096      // R + U + S
#define B_SZ    2
#define D_MODEL 1024
#define H_HEADS 16
#define HD      64
#define KK      1095      // M + (T - S)
#define TP      1120      // T padded so Mt=70 (even)
#define KP      1152      // Kk padded so Nt=72 (mult of 4) and K%32==0
#define BHN     32        // B * H
#define ROWS_TB 2192      // T * B (real rows)
#define ROWS_PAD 2208     // padded to 138 tiles (even)
#define ROWS_Y  2176      // (T - S) * B  (136 tiles, even)
#define FFN_DIM 4096
#define M_MEM   7
#define NPOS    129       // 2*MAXPOS + 1
#define NEG_INF (-__builtin_inff())

// ---------------- bf16 helpers ----------------
typedef __attribute__((ext_vector_type(16))) __bf16 v16bf;
typedef __attribute__((ext_vector_type(8)))  __bf16 v8bf;
typedef __attribute__((ext_vector_type(8)))  float  v8f;

__device__ __forceinline__ __bf16 f2bf(float f) {
    union { float f; unsigned u; } x; x.f = f;
    unsigned r = x.u + 0x7fffu + ((x.u >> 16) & 1u);   // round-to-nearest-even
    unsigned short h = (unsigned short)(r >> 16);
    return __builtin_bit_cast(__bf16, h);
}
__device__ __forceinline__ float bf2f(__bf16 b) {
    unsigned short h = __builtin_bit_cast(unsigned short, b);
    union { unsigned u; float f; } x; x.u = ((unsigned)h) << 16;
    return x.f;
}

union Frag { v16bf v; v8bf h[2]; };

// ---------------- WMMA GEMM: C[m,n] = sum_k A[m,k]*W[n,k] (+bias) ----------------
// Register-blocked: each wave computes a 32x64 tile (2 M-tiles x 4 N-tiles), so
// one K-step issues 12 b128 loads for 8 v_wmma_f32_16x16x32_bf16 (A reused 4x,
// B reused 2x). Fragment layout per CDNA5 ISA 7.12.2: lanes 0-15 hold K
// {0..7,16..23}, lanes 16-31 hold K {8..15,24..31}; C VGPR r: m=r+8*(lane>=16),
// n=lane%16. global_prefetch_b8 pulls the next K-panel ahead of the loads.
template<bool RELU, bool BF16OUT>
__global__ __launch_bounds__(32)
void gemm_wmma(const __bf16* __restrict__ A, long long sA,
               const __bf16* __restrict__ W, long long sW,
               void* __restrict__ Cv, long long sC,
               const float* __restrict__ bias, int K, int ldc)
{
    const int tn0 = blockIdx.x * 4, tm0 = blockIdx.y * 2, bz = blockIdx.z;
    const int lane = threadIdx.x;
    const int half = lane >> 4, lm = lane & 15;
    const __bf16* ap[2];
    ap[0] = A + (size_t)bz * sA + (size_t)(tm0 * 16 + lm) * K + half * 8;
    ap[1] = ap[0] + (size_t)16 * K;
    const __bf16* wp[4];
    wp[0] = W + (size_t)bz * sW + (size_t)(tn0 * 16 + lm) * K + half * 8;
    wp[1] = wp[0] + (size_t)16 * K;
    wp[2] = wp[1] + (size_t)16 * K;
    wp[3] = wp[2] + (size_t)16 * K;
    v8f acc[2][4] = {};
    for (int k = 0; k < K; k += 32) {
        if (k + 128 < K) {   // prefetch next K-panel (256B ahead per stream)
            __builtin_prefetch(ap[0] + k + 128, 0, 1);
            __builtin_prefetch(ap[1] + k + 128, 0, 1);
            __builtin_prefetch(wp[0] + k + 128, 0, 1);
            __builtin_prefetch(wp[1] + k + 128, 0, 1);
            __builtin_prefetch(wp[2] + k + 128, 0, 1);
            __builtin_prefetch(wp[3] + k + 128, 0, 1);
        }
        Frag a[2], b[4];
#pragma unroll
        for (int i = 0; i < 2; ++i) {
            a[i].h[0] = *(const v8bf*)(ap[i] + k);
            a[i].h[1] = *(const v8bf*)(ap[i] + k + 16);
        }
#pragma unroll
        for (int j = 0; j < 4; ++j) {
            b[j].h[0] = *(const v8bf*)(wp[j] + k);
            b[j].h[1] = *(const v8bf*)(wp[j] + k + 16);
        }
#pragma unroll
        for (int i = 0; i < 2; ++i)
#pragma unroll
            for (int j = 0; j < 4; ++j)
                acc[i][j] = __builtin_amdgcn_wmma_f32_16x16x32_bf16(
                    false, a[i].v, false, b[j].v, (short)0, acc[i][j], false, false);
    }
#pragma unroll
    for (int j = 0; j < 4; ++j) {
        const int n = (tn0 + j) * 16 + lm;
        const float bv = bias ? bias[n] : 0.f;
#pragma unroll
        for (int i = 0; i < 2; ++i) {
#pragma unroll
            for (int r = 0; r < 8; ++r) {
                int m = (tm0 + i) * 16 + r + half * 8;
                float v = acc[i][j][r] + bv;
                if (RELU) v = fmaxf(v, 0.f);
                size_t off = (size_t)bz * sC + (size_t)m * ldc + n;
                if (BF16OUT) ((__bf16*)Cv)[off] = f2bf(v);
                else         ((float*)Cv)[off]  = v;
            }
        }
    }
}

// ---------------- block reductions ----------------
__device__ __forceinline__ float blk_sum(float v, float* red) {
    int tid = threadIdx.x;
    red[tid] = v; __syncthreads();
    for (int o = blockDim.x >> 1; o > 0; o >>= 1) {
        if (tid < o) red[tid] += red[tid + o];
        __syncthreads();
    }
    float r = red[0]; __syncthreads();
    return r;
}
__device__ __forceinline__ float blk_max(float v, float* red) {
    int tid = threadIdx.x;
    red[tid] = v; __syncthreads();
    for (int o = blockDim.x >> 1; o > 0; o >>= 1) {
        if (tid < o) red[tid] = fmaxf(red[tid], red[tid + o]);
        __syncthreads();
    }
    float r = red[0]; __syncthreads();
    return r;
}

// ---------------- LayerNorm over D_MODEL, one row per block ----------------
__global__ __launch_bounds__(256)
void layernorm_k(const float* __restrict__ x, const float* __restrict__ res,
                 const float* __restrict__ w, const float* __restrict__ b,
                 float* __restrict__ out32, __bf16* __restrict__ outbf)
{
    __shared__ float red[256];
    const int row = blockIdx.x, tid = threadIdx.x;
    const float* xr = x + (size_t)row * D_MODEL;
    const float* rr = res ? res + (size_t)row * D_MODEL : nullptr;
    float v[4]; float s = 0.f;
#pragma unroll
    for (int j = 0; j < 4; ++j) {
        int c = tid + j * 256;
        float t = xr[c] + (rr ? rr[c] : 0.f);
        v[j] = t; s += t;
    }
    float mean = blk_sum(s, red) * (1.f / D_MODEL);
    float s2 = 0.f;
#pragma unroll
    for (int j = 0; j < 4; ++j) { float d = v[j] - mean; s2 += d * d; }
    float var = blk_sum(s2, red) * (1.f / D_MODEL);
    float inv = rsqrtf(var + 1e-5f);
#pragma unroll
    for (int j = 0; j < 4; ++j) {
        int c = tid + j * 256;
        float o = (v[j] - mean) * inv * w[c] + b[c];
        if (out32) out32[(size_t)row * D_MODEL + c] = o;
        if (outbf) outbf[(size_t)row * D_MODEL + c] = f2bf(o);
    }
}

// ---------------- elementwise ----------------
__global__ void cvt_bf16_k(const float* __restrict__ in, __bf16* __restrict__ out, size_t n) {
    size_t i = (size_t)blockIdx.x * blockDim.x + threadIdx.x;
    if (i < n) out[i] = f2bf(in[i]);
}

// zero the padded tail rows [ROWS_TB, ROWS_PAD) of hbf and attnc
__global__ void zero_pad_k(__bf16* __restrict__ hbf, __bf16* __restrict__ attnc) {
    int i = blockIdx.x * 256 + threadIdx.x;
    if (i < (ROWS_PAD - ROWS_TB) * D_MODEL) {
        size_t off = (size_t)ROWS_TB * D_MODEL + i;
        hbf[off]   = f2bf(0.f);
        attnc[off] = f2bf(0.f);
    }
}

// kv_in = concat(mems, h[:T-S]) flattened to (Kk*B, D), zero-padded to ROWS_PAD rows
__global__ void build_kvin_k(const float* __restrict__ mems, const __bf16* __restrict__ hbf,
                             __bf16* __restrict__ kvin) {
    size_t i = (size_t)blockIdx.x * blockDim.x + threadIdx.x;
    if (i >= (size_t)ROWS_PAD * D_MODEL) return;
    int r = (int)(i >> 10);
    __bf16 v;
    if (r < M_MEM * B_SZ)               v = f2bf(mems[i]);
    else if (r < M_MEM * B_SZ + ROWS_Y) v = hbf[i - (size_t)M_MEM * B_SZ * D_MODEL];
    else                                v = f2bf(0.f);
    kvin[i] = v;
}

// per-head padded repack: qh (scaled by hd^-0.5), kh, and V transposed (d-major)
__global__ void repack_k(const float* __restrict__ q32, const float* __restrict__ kv32,
                         __bf16* __restrict__ qh, __bf16* __restrict__ kh,
                         __bf16* __restrict__ vT) {
    size_t i = (size_t)blockIdx.x * blockDim.x + threadIdx.x;
    if (i >= (size_t)BHN * KP * HD) return;
    int d   = (int)(i % HD);
    int kkp = (int)((i / HD) % KP);
    int bh  = (int)(i / ((size_t)HD * KP));
    int b = bh >> 4, h = bh & 15;
    __bf16 kvv = f2bf(0.f), vvv = f2bf(0.f);
    if (kkp < KK) {
        const float* base = kv32 + (size_t)(kkp * B_SZ + b) * (2 * D_MODEL) + h * HD + d;
        kvv = f2bf(base[0]);
        vvv = f2bf(base[D_MODEL]);
    }
    kh[((size_t)bh * KP + kkp) * HD + d] = kvv;
    vT[((size_t)bh * HD + d) * KP + kkp] = vvv;
    if (kkp < TP) {
        __bf16 qv = f2bf(0.f);
        if (kkp < T_LEN)
            qv = f2bf(q32[(size_t)(kkp * B_SZ + b) * D_MODEL + h * HD + d] * 0.125f);
        qh[((size_t)bh * TP + kkp) * HD + d] = qv;
    }
}

// rpe score table: rpesc[bh][t][p] = dot(qh[bh,t,:], Ek[p,:])
__global__ __launch_bounds__(192)
void rpe_scores_k(const __bf16* __restrict__ qh, const float* __restrict__ Ek,
                  float* __restrict__ rpesc) {
    __shared__ float qrow[HD];
    const int t = blockIdx.x, bh = blockIdx.y, tid = threadIdx.x;
    if (tid < HD) qrow[tid] = bf2f(qh[((size_t)bh * TP + t) * HD + tid]);
    __syncthreads();
    if (tid < NPOS) {
        const float* e = Ek + tid * HD;
        float a = 0.f;
#pragma unroll 8
        for (int d = 0; d < HD; ++d) a += qrow[d] * e[d];
        rpesc[((size_t)bh * T_LEN + t) * NPOS + tid] = a;
    }
}

// scores += gathered rpe; apply attention_mask, key-length pad mask, col padding
__global__ void addmask_k(float* __restrict__ scores, const float* __restrict__ rpesc,
                          const int* __restrict__ rpe, const unsigned char* __restrict__ amask,
                          const int* __restrict__ lengths, const int* __restrict__ Rp) {
    size_t i = (size_t)blockIdx.x * blockDim.x + threadIdx.x;
    if (i >= (size_t)BHN * T_LEN * KP) return;
    int kkp = (int)(i % KP);
    size_t r = i / KP;
    int t = (int)(r % T_LEN);
    int bh = (int)(r / T_LEN);
    size_t so = ((size_t)bh * TP + t) * KP + kkp;
    float s = scores[so];
    bool dead = (kkp >= KK);
    if (!dead) {
        s += rpesc[((size_t)bh * T_LEN + t) * NPOS + rpe[(size_t)t * KK + kkp]];
        int b = bh >> 4;
        int klen = lengths[b] + M_MEM + Rp[0];
        dead = (amask[(size_t)t * KK + kkp] != 0) || (kkp >= klen);
    }
    scores[so] = dead ? NEG_INF : s;
}

// attention suppression: softmax stats in LDS, re-mask scores below threshold
__global__ __launch_bounds__(256)
void suppress_k(float* __restrict__ scores) {
    __shared__ float srow[KP];
    __shared__ float red[256];
    const int t = blockIdx.x, bh = blockIdx.y, tid = threadIdx.x;
    float* row = scores + ((size_t)bh * TP + t) * KP;
    for (int k = tid; k < KP; k += 256) srow[k] = row[k];
    __syncthreads();
    float mx = NEG_INF;
    for (int k = tid; k < KP; k += 256) mx = fmaxf(mx, srow[k]);
    mx = blk_max(mx, red);
    float se = 0.f, nn = 0.f;
    for (int k = tid; k < KP; k += 256) {
        float s = srow[k];
        if (s > -1e38f) { se += __expf(s - mx); nn += 1.f; }
    }
    float sumE = blk_sum(se, red);
    float nz   = blk_sum(nn, red);
    float km = 1.0f / (nz + 1e-8f);          // sum(prob) == 1
    float invE = 1.0f / sumE;
    float dis = 0.f;
    for (int k = tid; k < KP; k += 256) {
        float s = srow[k];
        if (s > -1e38f) { float d = __expf(s - mx) * invE - km; dis += d * d; }
    }
    float disT = blk_sum(dis, red);
    float sd  = sqrtf(disT / (nz - 1.f + 1e-8f));
    float thr = km - 0.5f * sd;
    for (int k = tid; k < KP; k += 256) {
        float s = srow[k];
        float p = (s > -1e38f) ? __expf(s - mx) * invE : 0.f;
        row[k] = (p < thr) ? NEG_INF : s;
    }
}

// final softmax -> bf16 probs
__global__ __launch_bounds__(256)
void softmax_probs_k(const float* __restrict__ scores, __bf16* __restrict__ probs) {
    __shared__ float srow[KP];
    __shared__ float red[256];
    const int t = blockIdx.x, bh = blockIdx.y, tid = threadIdx.x;
    const float* row = scores + ((size_t)bh * TP + t) * KP;
    for (int k = tid; k < KP; k += 256) srow[k] = row[k];
    __syncthreads();
    float mx = NEG_INF;
    for (int k = tid; k < KP; k += 256) mx = fmaxf(mx, srow[k]);
    mx = blk_max(mx, red);
    float se = 0.f;
    for (int k = tid; k < KP; k += 256) {
        float s = srow[k];
        if (s > -1e38f) se += __expf(s - mx);
    }
    float inv = 1.0f / blk_sum(se, red);
    __bf16* prow = probs + ((size_t)bh * TP + t) * KP;
    for (int k = tid; k < KP; k += 256) {
        float s = srow[k];
        prow[k] = f2bf((s > -1e38f) ? __expf(s - mx) * inv : 0.f);
    }
}

// attn = 2*(P@V) + sum_p bin(p)*Ev[p];  write as (T,B,D) bf16 for Wo GEMM
__global__ __launch_bounds__(128)
void combine_k(const __bf16* __restrict__ probs, const int* __restrict__ rpe,
               const float* __restrict__ Ev, const float* __restrict__ attnv,
               __bf16* __restrict__ attnc) {
    __shared__ float pw[NPOS];
    const int t = blockIdx.x, bh = blockIdx.y, tid = threadIdx.x;
    if (tid < NPOS) pw[tid] = 0.f;
    __syncthreads();
    const __bf16* prow = probs + ((size_t)bh * TP + t) * KP;
    for (int k = tid; k < KK; k += 128)
        atomicAdd(&pw[rpe[(size_t)t * KK + k]], bf2f(prow[k]));
    __syncthreads();
    if (tid < HD) {
        int d = tid;
        float rv = 0.f;
        for (int p = 0; p < NPOS; ++p) rv += pw[p] * Ev[p * HD + d];
        int b = bh >> 4, h = bh & 15;
        float val = 2.f * attnv[((size_t)bh * TP + t) * HD + d] + rv;
        attnc[(size_t)(t * B_SZ + b) * D_MODEL + h * HD + d] = f2bf(val);
    }
}

// y = out[:T-S] + src ; next_m = clip(out[T-S:]) straight into d_out
__global__ void resid_split_k(const float* __restrict__ out32, const float* __restrict__ x,
                              float* __restrict__ y32, float* __restrict__ dout) {
    size_t i = (size_t)blockIdx.x * blockDim.x + threadIdx.x;
    if (i >= (size_t)ROWS_TB * D_MODEL) return;
    int r = (int)(i >> 10);
    float o = out32[i];
    if (r < ROWS_Y) y32[i] = o + x[i];
    else            dout[i] = fminf(10.f, fmaxf(-10.f, o));  // flat offset == i
}

// ---------------- driver ----------------
extern "C" void kernel_launch(void* const* d_in, const int* in_sizes, int n_in,
                              void* d_out, int out_size, void* d_ws, size_t ws_size,
                              hipStream_t stream) {
    const float* x       = (const float*)d_in[0];
    const float* mems    = (const float*)d_in[1];
    const int*   lengths = (const int*)  d_in[2];
    const unsigned char* amask = (const unsigned char*)d_in[3];
    const int*   rpe     = (const int*)  d_in[4];
    const int*   Rp      = (const int*)  d_in[5];
    const float* Wq      = (const float*)d_in[6];
    const float* bq      = (const float*)d_in[7];
    const float* Wkv     = (const float*)d_in[8];
    const float* bkv     = (const float*)d_in[9];
    const float* Ek      = (const float*)d_in[10];
    const float* Ev      = (const float*)d_in[11];
    const float* Wo      = (const float*)d_in[12];
    const float* bo      = (const float*)d_in[13];
    const float* lnpw    = (const float*)d_in[14];
    const float* lnpb    = (const float*)d_in[15];
    const float* fflw    = (const float*)d_in[16];
    const float* fflb    = (const float*)d_in[17];
    const float* W1      = (const float*)d_in[18];
    const float* b1      = (const float*)d_in[19];
    const float* W2      = (const float*)d_in[20];
    const float* b2      = (const float*)d_in[21];
    const float* lnw     = (const float*)d_in[22];
    const float* lnb     = (const float*)d_in[23];
    float* dout = (float*)d_out;

    char* ws = (char*)d_ws;
    size_t off = 0;
    auto take = [&](size_t bytes) -> void* {
        void* p = ws + off;
        off = (off + bytes + 255) & ~(size_t)255;
        return p;
    };
    __bf16* hbf     = (__bf16*)take((size_t)ROWS_PAD * D_MODEL * 2);
    float*  q32     = (float*) take((size_t)ROWS_PAD * D_MODEL * 4);
    __bf16* kvinbf  = (__bf16*)take((size_t)ROWS_PAD * D_MODEL * 2);
    float*  kv32    = (float*) take((size_t)ROWS_PAD * 2 * D_MODEL * 4);
    __bf16* qh      = (__bf16*)take((size_t)BHN * TP * HD * 2);
    __bf16* kh      = (__bf16*)take((size_t)BHN * KP * HD * 2);
    __bf16* vT      = (__bf16*)take((size_t)BHN * HD * KP * 2);
    float*  rpesc   = (float*) take((size_t)BHN * T_LEN * NPOS * 4);
    float*  scores  = (float*) take((size_t)BHN * TP * KP * 4);
    __bf16* probs   = (__bf16*)take((size_t)BHN * TP * KP * 2);
    float*  attnv   = (float*) take((size_t)BHN * TP * HD * 4);
    __bf16* attnc   = (__bf16*)take((size_t)ROWS_PAD * D_MODEL * 2);
    float*  out32   = (float*) take((size_t)ROWS_PAD * D_MODEL * 4);
    float*  y32     = (float*) take((size_t)ROWS_Y * D_MODEL * 4);
    __bf16* ylnbf   = (__bf16*)take((size_t)ROWS_Y * D_MODEL * 2);
    __bf16* ff1bf   = (__bf16*)take((size_t)ROWS_Y * FFN_DIM * 2);
    float*  ff2     = (float*) take((size_t)ROWS_Y * D_MODEL * 4);
    __bf16* Wqbf    = (__bf16*)take((size_t)D_MODEL * D_MODEL * 2);
    __bf16* Wkvbf   = (__bf16*)take((size_t)2 * D_MODEL * D_MODEL * 2);
    __bf16* Wobf    = (__bf16*)take((size_t)D_MODEL * D_MODEL * 2);
    __bf16* W1bf    = (__bf16*)take((size_t)FFN_DIM * D_MODEL * 2);
    __bf16* W2bf    = (__bf16*)take((size_t)D_MODEL * FFN_DIM * 2);
    (void)ws_size; (void)in_sizes; (void)n_in; (void)out_size;

    auto blocks = [](size_t n) { return (unsigned)((n + 255) / 256); };

    // weight conversions
    cvt_bf16_k<<<blocks((size_t)D_MODEL * D_MODEL), 256, 0, stream>>>(Wq, Wqbf, (size_t)D_MODEL * D_MODEL);
    cvt_bf16_k<<<blocks((size_t)2 * D_MODEL * D_MODEL), 256, 0, stream>>>(Wkv, Wkvbf, (size_t)2 * D_MODEL * D_MODEL);
    cvt_bf16_k<<<blocks((size_t)D_MODEL * D_MODEL), 256, 0, stream>>>(Wo, Wobf, (size_t)D_MODEL * D_MODEL);
    cvt_bf16_k<<<blocks((size_t)FFN_DIM * D_MODEL), 256, 0, stream>>>(W1, W1bf, (size_t)FFN_DIM * D_MODEL);
    cvt_bf16_k<<<blocks((size_t)D_MODEL * FFN_DIM), 256, 0, stream>>>(W2, W2bf, (size_t)D_MODEL * FFN_DIM);

    // h = LN_pre(x); zero padded tail rows of hbf / attnc
    layernorm_k<<<ROWS_TB, 256, 0, stream>>>(x, nullptr, lnpw, lnpb, nullptr, hbf);
    zero_pad_k<<<blocks((size_t)(ROWS_PAD - ROWS_TB) * D_MODEL), 256, 0, stream>>>(hbf, attnc);
    build_kvin_k<<<blocks((size_t)ROWS_PAD * D_MODEL), 256, 0, stream>>>(mems, hbf, kvinbf);

    // Q and KV projections (WMMA, 32x64 per wave)
    gemm_wmma<false, false><<<dim3(D_MODEL / 64, ROWS_PAD / 32, 1), 32, 0, stream>>>(
        hbf, 0, Wqbf, 0, q32, 0, bq, D_MODEL, D_MODEL);
    gemm_wmma<false, false><<<dim3(2 * D_MODEL / 64, ROWS_PAD / 32, 1), 32, 0, stream>>>(
        kvinbf, 0, Wkvbf, 0, kv32, 0, bkv, D_MODEL, 2 * D_MODEL);

    repack_k<<<blocks((size_t)BHN * KP * HD), 256, 0, stream>>>(q32, kv32, qh, kh, vT);

    // scores = qh @ kh^T  (batched over 32 heads)
    gemm_wmma<false, false><<<dim3(KP / 64, TP / 32, BHN), 32, 0, stream>>>(
        qh, (long long)TP * HD, kh, (long long)KP * HD,
        scores, (long long)TP * KP, nullptr, HD, KP);

    rpe_scores_k<<<dim3(T_LEN, BHN), 192, 0, stream>>>(qh, Ek, rpesc);
    addmask_k<<<blocks((size_t)BHN * T_LEN * KP), 256, 0, stream>>>(
        scores, rpesc, rpe, amask, lengths, Rp);
    suppress_k<<<dim3(T_LEN, BHN), 256, 0, stream>>>(scores);
    softmax_probs_k<<<dim3(T_LEN, BHN), 256, 0, stream>>>(scores, probs);

    // attn = probs @ V (batched)
    gemm_wmma<false, false><<<dim3(HD / 64, TP / 32, BHN), 32, 0, stream>>>(
        probs, (long long)TP * KP, vT, (long long)HD * KP,
        attnv, (long long)TP * HD, nullptr, KP, HD);

    combine_k<<<dim3(T_LEN, BHN), 128, 0, stream>>>(probs, rpe, Ev, attnv, attnc);

    // output projection
    gemm_wmma<false, false><<<dim3(D_MODEL / 64, ROWS_PAD / 32, 1), 32, 0, stream>>>(
        attnc, 0, Wobf, 0, out32, 0, bo, D_MODEL, D_MODEL);

    resid_split_k<<<blocks((size_t)ROWS_TB * D_MODEL), 256, 0, stream>>>(out32, x, y32, dout);

    // FFN
    layernorm_k<<<ROWS_Y, 256, 0, stream>>>(y32, nullptr, fflw, fflb, nullptr, ylnbf);
    gemm_wmma<true, true><<<dim3(FFN_DIM / 64, ROWS_Y / 32, 1), 32, 0, stream>>>(
        ylnbf, 0, W1bf, 0, ff1bf, 0, b1, D_MODEL, FFN_DIM);
    gemm_wmma<false, false><<<dim3(D_MODEL / 64, ROWS_Y / 32, 1), 32, 0, stream>>>(
        ff1bf, 0, W2bf, 0, ff2, 0, b2, FFN_DIM, D_MODEL);

    // y = LN(y + ff) -> first part of d_out
    layernorm_k<<<ROWS_Y, 256, 0, stream>>>(ff2, y32, lnw, lnb, dout, nullptr);
}
